// TransformerMessageBlock_13005160972670
// MI455X (gfx1250) — matrix operand order repaired and checked by reading, not compiled, not run on hardware
//
#include <hip/hip_runtime.h>

#define N_NODES 10000
#define N_EDGES 250000
#define FEAT    128
#define HEADS   4
#define HF      512      // HEADS*FEAT
#define N_RBF   50
#define KRBF    64       // padded RBF K for WMMA
#define QKV_COLS 1536    // q|k|v concatenated
#define DKV_COLS 1024    // dk|dv concatenated
#define OUT_COLS 384     // 3*FEAT

typedef _Float16 half_t;
typedef __attribute__((ext_vector_type(16))) _Float16 v16h;
typedef __attribute__((ext_vector_type(8)))  _Float16 v8h;
typedef __attribute__((ext_vector_type(8)))  float    v8f;

__device__ __forceinline__ float silu_f(float x) {
    return x / (1.0f + __expf(-x));
}

// A-fragment (16x32 f16) from row-major LDS tile [16][ld]; two aligned b128 loads
__device__ __forceinline__ v16h load_A_lds(const half_t* base, int ld, int k0, int lane) {
    int m  = lane & 15;
    int kh = (lane >> 4) * 16;
    const half_t* row = base + m * ld + k0 + kh;
    v8h lo = *(const v8h*)(row);
    v8h hi = *(const v8h*)(row + 8);
    return __builtin_shufflevector(lo, hi, 0,1,2,3,4,5,6,7,8,9,10,11,12,13,14,15);
}

// B-fragment from pre-swizzled weights: fragment f = 32 lanes x 16 f16 contiguous
__device__ __forceinline__ v16h load_B_frag(const half_t* Wsw, int frag, int lane) {
    return *(const v16h*)(Wsw + ((size_t)frag * 32 + lane) * 16);
}

__device__ __forceinline__ v8f wmma_f16(v16h a, v16h b, v8f c) {
    return __builtin_amdgcn_wmma_f32_16x16x32_f16(false, a, false, b, (short)0, c, false, false);
}

// ---------------- weight prep: fp32 -> f16, fragment-swizzled ----------------
// dst element idx: frag = idx>>9, lane = (idx>>4)&31, e = idx&15
// fragment (nt,kt): frag = nt*numKt + kt ; source (k,n):
//   n = nt*16 + (lane&15), k = kt*32 + (lane>>4)*16 + e

__global__ void prep_wqkv_sw(const float* Wq, const float* Wk, const float* Wv, half_t* dst) {
    int idx = blockIdx.x * blockDim.x + threadIdx.x;
    if (idx >= FEAT * QKV_COLS) return;
    int e = idx & 15, lane = (idx >> 4) & 31, frag = idx >> 9;
    const int numKt = FEAT / 32;                 // 4
    int kt = frag % numKt, nt = frag / numKt;
    int n = nt * 16 + (lane & 15);
    int k = kt * 32 + (lane >> 4) * 16 + e;
    float v;
    if (n < HF)          v = Wq[k * HF + n];
    else if (n < 2 * HF) v = Wk[k * HF + (n - HF)];
    else                 v = Wv[k * HF + (n - 2 * HF)];
    dst[idx] = (half_t)v;
}

__global__ void prep_wdkv_sw(const float* Wdk, const float* Wdv, half_t* dst) {
    int idx = blockIdx.x * blockDim.x + threadIdx.x;
    if (idx >= KRBF * DKV_COLS) return;
    int e = idx & 15, lane = (idx >> 4) & 31, frag = idx >> 9;
    const int numKt = KRBF / 32;                 // 2
    int kt = frag % numKt, nt = frag / numKt;
    int n = nt * 16 + (lane & 15);
    int k = kt * 32 + (lane >> 4) * 16 + e;
    float v = 0.0f;
    if (k < N_RBF) v = (n < HF) ? Wdk[k * HF + n] : Wdv[k * HF + (n - HF)];
    dst[idx] = (half_t)v;
}

__global__ void prep_wd_sw(const float* Wd, half_t* dst) {
    int idx = blockIdx.x * blockDim.x + threadIdx.x;
    if (idx >= HF * OUT_COLS) return;
    int e = idx & 15, lane = (idx >> 4) & 31, frag = idx >> 9;
    const int numKt = HF / 32;                   // 16
    int kt = frag % numKt, nt = frag / numKt;
    int n = nt * 16 + (lane & 15);
    int k = kt * 32 + (lane >> 4) * 16 + e;
    dst[idx] = (half_t)Wd[k * OUT_COLS + n];
}

// ---------------- node kernel: LayerNorm + q/k/v projection ----------------
__global__ __launch_bounds__(128) void node_qkv_kernel(
    const float* __restrict__ s, const float* __restrict__ gamma, const float* __restrict__ beta,
    const half_t* __restrict__ Wqkv,
    const float* __restrict__ bq, const float* __restrict__ bk, const float* __restrict__ bv,
    half_t* __restrict__ qkv_out)
{
    __shared__ half_t sX[16 * FEAT];          // 4 KB
    __shared__ half_t sQKV[16 * QKV_COLS];    // 48 KB staging for coalesced stores
    const int tid = threadIdx.x, lane = tid & 31, wave = tid >> 5;
    const int n0 = blockIdx.x * 16;

    // LayerNorm: each wave owns 4 nodes; 32 lanes reduce 128 features
#pragma unroll
    for (int nn = 0; nn < 4; ++nn) {
        int node = wave * 4 + nn;
        const float* row = s + (size_t)(n0 + node) * FEAT;
        float v0 = row[lane], v1 = row[lane + 32], v2 = row[lane + 64], v3 = row[lane + 96];
        float sum = v0 + v1 + v2 + v3;
        float sq  = v0 * v0 + v1 * v1 + v2 * v2 + v3 * v3;
#pragma unroll
        for (int off = 16; off > 0; off >>= 1) {
            sum += __shfl_xor(sum, off, 32);
            sq  += __shfl_xor(sq,  off, 32);
        }
        float mu  = sum * (1.0f / FEAT);
        float var = sq * (1.0f / FEAT) - mu * mu;
        float inv = rsqrtf(var + 1e-5f);
        int c;
        c = lane;      sX[node * FEAT + c] = (half_t)((v0 - mu) * inv * gamma[c] + beta[c]);
        c = lane + 32; sX[node * FEAT + c] = (half_t)((v1 - mu) * inv * gamma[c] + beta[c]);
        c = lane + 64; sX[node * FEAT + c] = (half_t)((v2 - mu) * inv * gamma[c] + beta[c]);
        c = lane + 96; sX[node * FEAT + c] = (half_t)((v3 - mu) * inv * gamma[c] + beta[c]);
    }
    __syncthreads();

    // x(16x128) @ Wqkv(128x1536): 96 N-tiles, 24 per wave, 2-wide blocking, 4 K-steps
    const int numKt = FEAT / 32;   // 4
    for (int t = 0; t < 12; ++t) {
        int nt0 = wave * 24 + t * 2;
        v8f acc0 = {}, acc1 = {};
#pragma unroll
        for (int kt = 0; kt < numKt; ++kt) {
            v16h a  = load_A_lds(sX, FEAT, kt * 32, lane);
            v16h b0 = load_B_frag(Wqkv, nt0 * numKt + kt, lane);
            v16h b1 = load_B_frag(Wqkv, (nt0 + 1) * numKt + kt, lane);
            acc0 = wmma_f16(a, b0, acc0);
            acc1 = wmma_f16(a, b1, acc1);
        }
        int mbase = 8 * (lane >> 4);
#pragma unroll
        for (int half = 0; half < 2; ++half) {
            v8f* accp = half ? &acc1 : &acc0;
            int c = (nt0 + half) * 16 + (lane & 15);
            float bias = (c < HF) ? bq[c] : (c < 2 * HF) ? bk[c - HF] : bv[c - 2 * HF];
#pragma unroll
            for (int r = 0; r < 8; ++r)
                sQKV[(mbase + r) * QKV_COLS + c] = (half_t)((*accp)[r] + bias);
        }
    }
    __syncthreads();

    // coalesced b128 stores: 16 rows x 1536 halves
    for (int idx = tid; idx < 16 * QKV_COLS / 8; idx += 128) {
        int m = idx / (QKV_COLS / 8);
        int c = (idx % (QKV_COLS / 8)) * 8;
        *(v8h*)(qkv_out + (size_t)(n0 + m) * QKV_COLS + c) = *(const v8h*)(sQKV + m * QKV_COLS + c);
    }
}

// ---------------- fused edge kernel ----------------
__global__ __launch_bounds__(128) void edge_kernel(
    const float* __restrict__ r_ij, const int* __restrict__ nbrs, const float* __restrict__ v_j,
    const half_t* __restrict__ Wdkv, const float* __restrict__ bdk, const float* __restrict__ bdv,
    const half_t* __restrict__ Wd_h, const float* __restrict__ bd,
    const half_t* __restrict__ qkv,
    float* __restrict__ out_s, float* __restrict__ out_v)
{
    __shared__ __align__(16) char smem[49152];
    half_t* sDkv = (half_t*)smem;                 // [16][1024] dk|dv (post-silu)
    half_t* sMsg = (half_t*)(smem + 32768);       // [16][512] msg; earlier aliases sA
    half_t* sA   = sMsg;                          // [16][64] padded rbf tile
    float*  sOut = (float*)smem;                  // [16][384], reuses sDkv region
    __shared__ float sAttn[16 * HEADS];
    __shared__ float sUnit[16 * 3];
    __shared__ float sDist[16];
    __shared__ int   sI[16], sJ[16];

    const int tid = threadIdx.x, lane = tid & 31, wave = tid >> 5;
    const int e0 = blockIdx.x * 16;

    // stage 1a: geometry + indices
    if (tid < 16) {
        int e = e0 + tid;
        float rx = r_ij[e * 3], ry = r_ij[e * 3 + 1], rz = r_ij[e * 3 + 2];
        float d = sqrtf(rx * rx + ry * ry + rz * rz + 3.0f * 1e-15f);
        sDist[tid] = d;
        float invd = 1.0f / d;
        sUnit[tid * 3]     = rx * invd;
        sUnit[tid * 3 + 1] = ry * invd;
        sUnit[tid * 3 + 2] = rz * invd;
        sI[tid] = nbrs[e * 2];
        sJ[tid] = nbrs[e * 2 + 1];
    }
    __syncthreads();

    // stage 1b: RBF expansion (K padded to 64 with zeros)
    const float width = (float)(5.0 / (N_RBF - 1));
    const float grbf  = 0.5f / (width * width);
    for (int idx = tid; idx < 16 * KRBF; idx += 128) {
        int e = idx >> 6, kk = idx & 63;
        float val = 0.0f;
        if (kk < N_RBF) {
            float dd = sDist[e] - width * (float)kk;
            val = __expf(-grbf * dd * dd);
        }
        sA[idx] = (half_t)val;
    }
    __syncthreads();

    // stage 2: dk|dv = silu(rbf @ Wdkv + b): 64 N-tiles, 16/wave, 2-wide, 2 K-steps
    {
        const int numKt = KRBF / 32;   // 2
        for (int t = 0; t < 8; ++t) {
            int nt0 = wave * 16 + t * 2;
            v8f acc0 = {}, acc1 = {};
#pragma unroll
            for (int kt = 0; kt < numKt; ++kt) {
                v16h a  = load_A_lds(sA, KRBF, kt * 32, lane);
                v16h b0 = load_B_frag(Wdkv, nt0 * numKt + kt, lane);
                v16h b1 = load_B_frag(Wdkv, (nt0 + 1) * numKt + kt, lane);
                acc0 = wmma_f16(a, b0, acc0);
                acc1 = wmma_f16(a, b1, acc1);
            }
            int mbase = 8 * (lane >> 4);
#pragma unroll
            for (int half = 0; half < 2; ++half) {
                v8f* accp = half ? &acc1 : &acc0;
                int c = (nt0 + half) * 16 + (lane & 15);
                float bias = (c < HF) ? bdk[c] : bdv[c - HF];
#pragma unroll
                for (int r = 0; r < 8; ++r)
                    sDkv[(mbase + r) * DKV_COLS + c] = (half_t)silu_f((*accp)[r] + bias);
            }
        }
    }
    __syncthreads();

    // stage 3: attn = silu(sum_f q[i]*k[j]*dk), fp32, vectorized gathers
    if (tid < 16 * HEADS) {
        int e = tid >> 2, h = tid & 3;
        const half_t* qrow  = qkv + (size_t)sI[e] * QKV_COLS + h * FEAT;
        const half_t* krow  = qkv + (size_t)sJ[e] * QKV_COLS + HF + h * FEAT;
        const half_t* dkrow = sDkv + e * DKV_COLS + h * FEAT;
        float acc = 0.0f;
#pragma unroll
        for (int f = 0; f < FEAT; f += 8) {
            v8h qv = *(const v8h*)(qrow + f);
            v8h kv = *(const v8h*)(krow + f);
            v8h dk = *(const v8h*)(dkrow + f);
#pragma unroll
            for (int u = 0; u < 8; ++u)
                acc += (float)qv[u] * (float)kv[u] * (float)dk[u];
        }
        sAttn[e * HEADS + h] = silu_f(acc);
    }
    __syncthreads();

    // stage 4: msg = v[j] * dv * attn -> f16 tile [16][512], 8-wide
    for (int idx = tid; idx < 16 * HF / 8; idx += 128) {
        int e = idx >> 6;
        int c = (idx & 63) * 8;
        int h = c >> 7;
        v8h vv = *(const v8h*)(qkv + (size_t)sJ[e] * QKV_COLS + 2 * HF + c);
        v8h dv = *(const v8h*)(sDkv + e * DKV_COLS + HF + c);
        float at = sAttn[e * HEADS + h];
        v8h m;
#pragma unroll
        for (int u = 0; u < 8; ++u)
            m[u] = (half_t)((float)vv[u] * (float)dv[u] * at);
        *(v8h*)(sMsg + e * HF + c) = m;
    }
    __syncthreads();

    // stage 5: out = msg(16x512) @ Wd(512x384) + bd: 24 N-tiles, 6/wave, 2-wide, 16 K-steps
    {
        const int numKt = HF / 32;   // 16
        for (int t = 0; t < 3; ++t) {
            int nt0 = wave * 6 + t * 2;
            v8f acc0 = {}, acc1 = {};
#pragma unroll
            for (int kt = 0; kt < numKt; ++kt) {
                v16h a  = load_A_lds(sMsg, HF, kt * 32, lane);
                v16h b0 = load_B_frag(Wd_h, nt0 * numKt + kt, lane);
                v16h b1 = load_B_frag(Wd_h, (nt0 + 1) * numKt + kt, lane);
                acc0 = wmma_f16(a, b0, acc0);
                acc1 = wmma_f16(a, b1, acc1);
            }
            int mbase = 8 * (lane >> 4);
#pragma unroll
            for (int half = 0; half < 2; ++half) {
                v8f* accp = half ? &acc1 : &acc0;
                int c = (nt0 + half) * 16 + (lane & 15);
                float bias = bd[c];
#pragma unroll
                for (int r = 0; r < 8; ++r)
                    sOut[(mbase + r) * OUT_COLS + c] = (*accp)[r] + bias;  // sDkv dead now
            }
        }
    }
    __syncthreads();

    // stage 6: scatter-add into node outputs (fp32 atomics, L2-resident)
    for (int idx = tid; idx < 16 * FEAT; idx += 128) {
        int e = idx >> 7, f = idx & 127;
        int i = sI[e], j = sJ[e];
        float s0 = sOut[e * OUT_COLS + f];               // gates v_j
        float s1 = sOut[e * OUT_COLS + FEAT + f];        // delta_s
        float s2 = sOut[e * OUT_COLS + 2 * FEAT + f];    // gates unit
        atomicAdd(&out_s[(size_t)i * FEAT + f], s1);
        const float* vj = v_j + ((size_t)j * FEAT + f) * 3;
        float* dv = out_v + ((size_t)i * FEAT + f) * 3;
#pragma unroll
        for (int d = 0; d < 3; ++d)
            atomicAdd(&dv[d], s2 * sUnit[e * 3 + d] + s0 * vj[d]);
    }
}

extern "C" void kernel_launch(void* const* d_in, const int* in_sizes, int n_in,
                              void* d_out, int out_size, void* d_ws, size_t ws_size,
                              hipStream_t stream) {
    (void)in_sizes; (void)n_in; (void)ws_size;
    const float* s_j      = (const float*)d_in[0];
    const float* v_j      = (const float*)d_in[1];
    const float* r_ij     = (const float*)d_in[2];
    const int*   nbrs     = (const int*)d_in[3];
    const float* ln_gamma = (const float*)d_in[4];
    const float* ln_beta  = (const float*)d_in[5];
    const float* Wq  = (const float*)d_in[6];   const float* bq  = (const float*)d_in[7];
    const float* Wk  = (const float*)d_in[8];   const float* bk  = (const float*)d_in[9];
    const float* Wv  = (const float*)d_in[10];  const float* bv  = (const float*)d_in[11];
    const float* Wdk = (const float*)d_in[12];  const float* bdk = (const float*)d_in[13];
    const float* Wdv = (const float*)d_in[14];  const float* bdv = (const float*)d_in[15];
    const float* Wd  = (const float*)d_in[16];  const float* bd  = (const float*)d_in[17];

    char* ws = (char*)d_ws;
    half_t* Wqkv_h = (half_t*)ws;  ws += (size_t)FEAT * QKV_COLS * sizeof(half_t);
    half_t* Wdkv_h = (half_t*)ws;  ws += (size_t)KRBF * DKV_COLS * sizeof(half_t);
    half_t* Wd_h   = (half_t*)ws;  ws += (size_t)HF * OUT_COLS * sizeof(half_t);
    half_t* qkv_h  = (half_t*)ws;  // N_NODES * 1536 f16 (~30.7 MB)

    float* out_s = (float*)d_out;
    float* out_v = out_s + (size_t)N_NODES * FEAT;

    hipMemsetAsync(d_out, 0, (size_t)out_size * sizeof(float), stream);

    prep_wqkv_sw<<<(FEAT * QKV_COLS + 255) / 256, 256, 0, stream>>>(Wq, Wk, Wv, Wqkv_h);
    prep_wdkv_sw<<<(KRBF * DKV_COLS + 255) / 256, 256, 0, stream>>>(Wdk, Wdv, Wdkv_h);
    prep_wd_sw<<<(HF * OUT_COLS + 255) / 256, 256, 0, stream>>>(Wd, Wd_h);

    node_qkv_kernel<<<N_NODES / 16, 128, 0, stream>>>(
        s_j, ln_gamma, ln_beta, Wqkv_h, bq, bk, bv, qkv_h);

    edge_kernel<<<N_EDGES / 16, 128, 0, stream>>>(
        r_ij, nbrs, v_j, Wdkv_h, bdk, bdv, Wd_h, bd, qkv_h, out_s, out_v);
}
